// Contextualized_Reasoner_Full_429496730144
// MI455X (gfx1250) — compile-verified, matrix operand order checked by reading, not run on hardware
//
#include <hip/hip_runtime.h>
#include <hip/hip_bf16.h>

// ---------------------------------------------------------------------------
// Contextualized reasoner forward, CDNA5 (gfx1250) implementation.
// All dense matmuls -> v_wmma_f32_16x16x32_bf16 (wave32, 128x64 block tile,
// K stepped by 64 through LDS). Weights pre-transposed to K-major bf16 so
// both LDS staging streams are contiguous b128 copies; staging goes through
// GLOBAL_LOAD_ASYNC_TO_LDS_B128 (ASYNCcnt) via inline asm. fp32 epilogues
// elsewhere.
// ---------------------------------------------------------------------------

#define B_   128
#define R_   6
#define C_   2048
#define HW_  49
#define D_   1024
#define E_   300
#define NH_  4
#define DH_  256            // D_/NH_
#define N_   (B_*R_)        // 768
#define NLBL 2001
#define K_QC   640          // 2E=600 padded to mult of 64
#define K_UQC  1664         // D+2E=1624 padded to mult of 64
#define NCLS_P 2048         // 2001 padded to mult of 64

// Async global->LDS staging (CDNA5 ASYNCcnt path). Set to 0 to fall back to
// plain vector loads if the assembler rejects the mnemonic/operands.
#define USE_ASYNC_LDS 1

typedef __attribute__((ext_vector_type(16))) __bf16 v16bf;
typedef __attribute__((ext_vector_type(8)))  float  v8f;

union FragAB { v16bf v; unsigned int u[8]; };

__device__ __forceinline__ unsigned short f2b(float f) {
    unsigned int u = __float_as_uint(f);
    unsigned int r = u + 0x7FFFu + ((u >> 16) & 1u);   // round-to-nearest-even
    return (unsigned short)(r >> 16);
}
__device__ __forceinline__ float b2f(unsigned short h) {
    return __uint_as_float(((unsigned int)h) << 16);
}
__device__ __forceinline__ float sigmoidf_(float x) { return 1.f / (1.f + expf(-x)); }

#if USE_ASYNC_LDS
// GLOBAL_LOAD_ASYNC_TO_LDS_B128: VDST = per-lane LDS byte address,
// VADDR = per-lane 64-bit global address, no SADDR. Tracked by ASYNCcnt.
__device__ __forceinline__ void async_copy_b128(const unsigned int* g, unsigned int* l) {
    unsigned int lds = (unsigned int)(size_t)l;          // addr[31:0] = LDS offset
    unsigned long long ga = (unsigned long long)(size_t)g;
    asm volatile("global_load_async_to_lds_b128 %0, %1, off"
                 :: "v"(lds), "v"(ga) : "memory");
}
__device__ __forceinline__ void wait_async0() {
    asm volatile("s_wait_asynccnt 0" ::: "memory");
}
#endif

// ---------------------------------------------------------------------------
// bf16 WMMA GEMM:  C[M x actN] = A[M x K] * B[K x N] (+bias)(+relu)
// A: [M x K] row-major bf16.  BT: [N x K] row-major bf16 (i.e. B transposed,
// K-contiguous). M % 128 == 0, N % 64 == 0, K % 64 == 0. actN clips stores.
// 256 threads = 8 waves; wave w owns a 16-row stripe and all 64 columns.
// ---------------------------------------------------------------------------
__global__ __launch_bounds__(256)
void k_gemm_bf16(const unsigned short* __restrict__ A,
                 const unsigned short* __restrict__ BT,
                 const float* __restrict__ bias,
                 float* __restrict__ Cm,
                 int M, int N, int K, int ldc, int actN, int epi)
{
    __shared__ unsigned int lA[128 * 32];   // 128 rows x 64 bf16 (row-major)
    __shared__ unsigned int lB[64 * 32];    // 64 cols  x 64 bf16 (K-contig)

    const int tid  = threadIdx.x;
    const int wv   = tid >> 5;
    const int lane = tid & 31;
    const int half = lane >> 4;
    const int l16  = lane & 15;
    const int mblk = blockIdx.y * 128;
    const int nblk = blockIdx.x * 64;

    const unsigned int* Au  = (const unsigned int*)A;
    const unsigned int* BTu = (const unsigned int*)BT;
    const int Ku = K >> 1;                  // uints per row

    // fixed per-thread staging assignment (all pointers hoisted out of K loop)
    const int arow = tid >> 1;              // 0..127
    const int acol = (tid & 1) * 16;        // uint offset within 32-uint row
    const unsigned int* aSrc = Au + (size_t)(mblk + arow) * Ku + acol;
    unsigned int*       aDst = &lA[arow * 32 + acol];

    const int bcol = tid >> 2;              // 0..63
    const int bq   = (tid & 3) * 8;         // uint offset within 32-uint col
    const unsigned int* bSrc = BTu + (size_t)(nblk + bcol) * Ku + bq;
    unsigned int*       bDst = &lB[bcol * 32 + bq];

    v8f zero = {0.f,0.f,0.f,0.f,0.f,0.f,0.f,0.f};
    v8f acc[4];
    acc[0] = zero; acc[1] = zero; acc[2] = zero; acc[3] = zero;

    for (int kk = 0; kk < K; kk += 64) {
        const int ko = kk >> 1;
        __syncthreads();
#if USE_ASYNC_LDS
        async_copy_b128(aSrc + ko,      aDst);
        async_copy_b128(aSrc + ko + 4,  aDst + 4);
        async_copy_b128(aSrc + ko + 8,  aDst + 8);
        async_copy_b128(aSrc + ko + 12, aDst + 12);
        async_copy_b128(bSrc + ko,      bDst);
        async_copy_b128(bSrc + ko + 4,  bDst + 4);
        wait_async0();
#else
        {
            const uint4* s = (const uint4*)(aSrc + ko);
            uint4*       d = (uint4*)aDst;
            d[0] = s[0]; d[1] = s[1]; d[2] = s[2]; d[3] = s[3];
            const uint4* sb = (const uint4*)(bSrc + ko);
            uint4*       db = (uint4*)bDst;
            db[0] = sb[0]; db[1] = sb[1];
        }
#endif
        if (kk + 64 < K) {   // prefetch next K tile (global_prefetch_b8)
            __builtin_prefetch(aSrc + ko + 32, 0, 1);
            __builtin_prefetch(bSrc + ko + 32, 0, 1);
        }
        __syncthreads();

        const unsigned int* ar = &lA[(wv * 16 + l16) * 32];
#pragma unroll
        for (int s = 0; s < 2; ++s) {
            // A fragment (16x32 bf16; lane m = l16; ISA K striping)
            FragAB fa;
            const unsigned int* ap = ar + s * 16;
#pragma unroll
            for (int j = 0; j < 4; ++j) fa.u[j]     = ap[4 * half + j];
#pragma unroll
            for (int j = 0; j < 4; ++j) fa.u[4 + j] = ap[8 + 4 * half + j];
#pragma unroll
            for (int t = 0; t < 4; ++t) {
                FragAB fb;
                const unsigned int* bp = &lB[(t * 16 + l16) * 32 + s * 16];
#pragma unroll
                for (int j = 0; j < 4; ++j) fb.u[j]     = bp[4 * half + j];
#pragma unroll
                for (int j = 0; j < 4; ++j) fb.u[4 + j] = bp[8 + 4 * half + j];
                acc[t] = __builtin_amdgcn_wmma_f32_16x16x32_bf16(
                    false, fa.v, false, fb.v, (short)0, acc[t], false, false);
            }
        }
    }

    // epilogue: C layout VGPR r -> m = r + 8*half, n = l16
#pragma unroll
    for (int t = 0; t < 4; ++t) {
        const int n = nblk + t * 16 + l16;
        if (n >= actN) continue;
        const float bv = bias ? bias[n] : 0.f;
#pragma unroll
        for (int r = 0; r < 8; ++r) {
            const int m = mblk + wv * 16 + r + 8 * half;
            if (m < M) {
                float val = acc[t][r] + bv;
                if (epi == 1) val = fmaxf(val, 0.f);
                Cm[(size_t)m * ldc + n] = val;
            }
        }
    }
}

// ---------------------------------------------------------------------------
// Helper kernels
// ---------------------------------------------------------------------------

// v_org [B,C,HW] -> vT bf16 [B*HW, C]
__global__ void k_pack_vT(const float* __restrict__ v, unsigned short* __restrict__ out, int total) {
    int i = blockIdx.x * 256 + threadIdx.x;
    if (i >= total) return;
    int c = i & (C_ - 1);
    int row = i >> 11;                 // /2048
    int b = row / HW_, hw = row % HW_;
    out[i] = f2b(v[((size_t)b * C_ + c) * HW_ + hw]);
}

// f32 [srows x scols] -> bf16 [drows x dcols], zero-padded (row-major, A side)
__global__ void k_pack(const float* __restrict__ src, unsigned short* __restrict__ dst,
                       int srows, int scols, int drows, int dcols) {
    int i = blockIdx.x * 256 + threadIdx.x;
    int total = drows * dcols;
    if (i >= total) return;
    int r = i / dcols, c = i % dcols;
    float v = (r < srows && c < scols) ? src[(size_t)r * scols + c] : 0.f;
    dst[i] = f2b(v);
}

// f32 B [srows x scols] -> bf16 BT [Ncols x Kpad] (K-contiguous), zero-padded
__global__ void k_packT(const float* __restrict__ src, unsigned short* __restrict__ dst,
                        int srows, int scols, int Kpad, int Ncols) {
    int i = blockIdx.x * 256 + threadIdx.x;
    int total = Ncols * Kpad;
    if (i >= total) return;
    int c = i / Kpad, r = i % Kpad;
    float v = (r < srows && c < scols) ? src[(size_t)r * scols + c] : 0.f;
    dst[i] = f2b(v);
}

// role_verb f32 [N, 2E] = cat(verb_emb[gt_verb[b]], role_emb[role_ids[gt_verb[b]][r]])
__global__ void k_roleverb(const float* __restrict__ verb_emb, const float* __restrict__ role_emb,
                           const int* __restrict__ role_ids, const int* __restrict__ gt_verb,
                           float* __restrict__ out, int total) {
    int i = blockIdx.x * 256 + threadIdx.x;
    if (i >= total) return;
    int n = i / (2 * E_), col = i % (2 * E_);
    int b = n / R_, r = n % R_;
    int verb = gt_verb[b];
    float v;
    if (col < E_) v = verb_emb[(size_t)verb * E_ + col];
    else {
        int rid = role_ids[verb * R_ + r];
        v = role_emb[(size_t)rid * E_ + (col - E_)];
    }
    out[i] = v;
}

// score[idx] = sum_d relu(att_v[vrow,d] + att_q[n,d]) * w_att[d]; idx = n*HW+hw
__global__ __launch_bounds__(256)
void k_score(const float* __restrict__ av, const float* __restrict__ aq,
             const float* __restrict__ w, float* __restrict__ sc, int sharedImg) {
    int idx = blockIdx.x;
    int n = idx / HW_, hw = idx % HW_, b = n / R_;
    size_t vrow = sharedImg ? ((size_t)b * HW_ + hw) : (size_t)idx;
    const float* vp = av + vrow * D_;
    const float* qp = aq + (size_t)n * D_;
    float s = 0.f;
    for (int d = threadIdx.x; d < D_; d += 256)
        s += fmaxf(vp[d] + qp[d], 0.f) * w[d];
    __shared__ float red[256];
    red[threadIdx.x] = s; __syncthreads();
    for (int st = 128; st > 0; st >>= 1) {
        if (threadIdx.x < st) red[threadIdx.x] += red[threadIdx.x + st];
        __syncthreads();
    }
    if (threadIdx.x == 0) sc[idx] = red[0];
}

// softmax over HW then weighted pool: v_emb[n,d] = sum_hw p[hw]*img[row(hw),d]
__global__ __launch_bounds__(256)
void k_pool(const float* __restrict__ sc, const float* __restrict__ imgf,
            const unsigned short* __restrict__ imgb, float* __restrict__ ve, int sharedImg) {
    int n = blockIdx.x, b = n / R_;
    __shared__ float p[HW_];
    if (threadIdx.x == 0) {
        float t[HW_]; float m = -1e30f;
        for (int h = 0; h < HW_; ++h) { t[h] = sc[n * HW_ + h]; m = fmaxf(m, t[h]); }
        float s = 0.f;
        for (int h = 0; h < HW_; ++h) { t[h] = expf(t[h] - m); s += t[h]; }
        float inv = 1.f / s;
        for (int h = 0; h < HW_; ++h) p[h] = t[h] * inv;
    }
    __syncthreads();
    for (int d = threadIdx.x; d < D_; d += 256) {
        float a = 0.f;
        for (int h = 0; h < HW_; ++h) {
            size_t row = sharedImg ? ((size_t)b * HW_ + h) : ((size_t)n * HW_ + h);
            float x = imgf ? imgf[row * D_ + d] : b2f(imgb[row * D_ + d]);
            a += p[h] * x;
        }
        ve[(size_t)n * D_ + d] = a;
    }
}

// out0[n,:] = l2norm(signed_sqrt(q*v))
__global__ __launch_bounds__(256)
void k_mfb(const float* __restrict__ q, const float* __restrict__ v, float* __restrict__ out) {
    int n = blockIdx.x, tid = threadIdx.x;
    float loc[4]; float ss = 0.f;
#pragma unroll
    for (int j = 0; j < 4; ++j) {
        size_t ix = (size_t)n * D_ + tid + j * 256;
        float x = q[ix] * v[ix];
        float t = x > 0.f ? sqrtf(x) : -sqrtf(-x);
        loc[j] = t; ss += t * t;
    }
    __shared__ float red[256];
    red[tid] = ss; __syncthreads();
    for (int st = 128; st > 0; st >>= 1) { if (tid < st) red[tid] += red[tid + st]; __syncthreads(); }
    __shared__ float invs;
    if (tid == 0) invs = 1.f / fmaxf(sqrtf(red[0]), 1e-12f);
    __syncthreads();
    float inv = invs;
#pragma unroll
    for (int j = 0; j < 4; ++j)
        out[(size_t)n * D_ + tid + j * 256] = loc[j] * inv;
}

// masked multi-head attention over R=6 roles; block per (b,h); 256 threads = dh
__global__ __launch_bounds__(256)
void k_roleattn(const float* __restrict__ Q, const float* __restrict__ Km,
                const float* __restrict__ V, float* __restrict__ ctx) {
    int b = blockIdx.x >> 2;
    int h = blockIdx.x & 3;
    int tid = threadIdx.x;
    __shared__ float pr[R_ * R_];
    if (tid < R_ * R_) {
        int i = tid / R_, j = tid % R_;
        const float* qi = Q  + (size_t)(b * R_ + i) * D_ + h * DH_;
        const float* kj = Km + (size_t)(b * R_ + j) * D_ + h * DH_;
        float s = 0.f;
        for (int d = 0; d < DH_; ++d) s += qi[d] * kj[d];
        pr[tid] = s * 0.0625f;   // 1/sqrt(256)
    }
    __syncthreads();
    if (tid == 0) {
        for (int i = 0; i < R_; ++i) {
            float m = -1e30f;
            for (int j = 0; j < R_; ++j) {
                float x = (i == j) ? -1e9f : pr[i * R_ + j];
                pr[i * R_ + j] = x; m = fmaxf(m, x);
            }
            float s = 0.f;
            for (int j = 0; j < R_; ++j) { pr[i * R_ + j] = expf(pr[i * R_ + j] - m); s += pr[i * R_ + j]; }
            float inv = 1.f / s;
            for (int j = 0; j < R_; ++j) pr[i * R_ + j] *= inv;
        }
    }
    __syncthreads();
    int d = tid;   // 256 == DH_
    for (int i = 0; i < R_; ++i) {
        float a = 0.f;
        for (int j = 0; j < R_; ++j)
            a += pr[i * R_ + j] * V[(size_t)(b * R_ + j) * D_ + h * DH_ + d];
        ctx[(size_t)(b * R_ + i) * D_ + h * DH_ + d] = a;
    }
}

// updated_img (bf16) = sigmoid(gA[n,d] + gB[b,hw,d] + b_ctx[d]) * ref[b,hw,d]
__global__ void k_gate(const float* __restrict__ gA, const float* __restrict__ gB,
                       const float* __restrict__ bctx, const float* __restrict__ ref,
                       unsigned short* __restrict__ uimg, int total) {
    int i = blockIdx.x * 256 + threadIdx.x;
    if (i >= total) return;
    int d = i & (D_ - 1);
    int t = i >> 10;                   // n*HW + hw
    int n = t / HW_, hw = t % HW_;
    int b = n / R_;
    size_t bi = ((size_t)b * HW_ + hw) * D_ + d;
    float g = sigmoidf_(gA[(size_t)n * D_ + d] + gB[bi] + bctx[d]);
    uimg[i] = f2b(g * ref[bi]);
}

// cat(bf16): [withctx(1024) | role_verb(600) | pad(40)]
__global__ void k_cat(const float* __restrict__ wctx, const float* __restrict__ rv,
                      unsigned short* __restrict__ out, int total) {
    int i = blockIdx.x * 256 + threadIdx.x;
    if (i >= total) return;
    int n = i / K_UQC, c = i % K_UQC;
    float v = (c < D_) ? wctx[(size_t)n * D_ + c]
            : (c < D_ + 2 * E_) ? rv[(size_t)n * (2 * E_) + (c - D_)] : 0.f;
    out[i] = f2b(v);
}

// out_t = l2norm(signed_sqrt(q2*v2)); gate = sigmoid(q1*q2);
// outb(bf16) = gate*out0 + (1-gate)*out_t
__global__ __launch_bounds__(256)
void k_final(const float* __restrict__ q1, const float* __restrict__ q2,
             const float* __restrict__ v2, const float* __restrict__ out0,
             unsigned short* __restrict__ outb) {
    int n = blockIdx.x, tid = threadIdx.x;
    float loc[4]; float ss = 0.f;
#pragma unroll
    for (int j = 0; j < 4; ++j) {
        size_t ix = (size_t)n * D_ + tid + j * 256;
        float x = q2[ix] * v2[ix];
        float t = x > 0.f ? sqrtf(x) : -sqrtf(-x);
        loc[j] = t; ss += t * t;
    }
    __shared__ float red[256];
    red[tid] = ss; __syncthreads();
    for (int st = 128; st > 0; st >>= 1) { if (tid < st) red[tid] += red[tid + st]; __syncthreads(); }
    __shared__ float invs;
    if (tid == 0) invs = 1.f / fmaxf(sqrtf(red[0]), 1e-12f);
    __syncthreads();
    float inv = invs;
#pragma unroll
    for (int j = 0; j < 4; ++j) {
        size_t ix = (size_t)n * D_ + tid + j * 256;
        float ot = loc[j] * inv;
        float g = sigmoidf_(q1[ix] * q2[ix]);
        outb[ix] = f2b(g * out0[ix] + (1.f - g) * ot);
    }
}

// ---------------------------------------------------------------------------
// Host launcher
// ---------------------------------------------------------------------------
static inline int ceilDiv(int a, int b) { return (a + b - 1) / b; }

extern "C" void kernel_launch(void* const* d_in, const int* in_sizes, int n_in,
                              void* d_out, int out_size, void* d_ws, size_t ws_size,
                              hipStream_t stream) {
    const float* v_org    = (const float*)d_in[0];
    const float* verb_emb = (const float*)d_in[1];
    const float* role_emb = (const float*)d_in[2];
    const float* W_ref    = (const float*)d_in[3];
    const float* b_ref    = (const float*)d_in[4];
    const float* W_qc     = (const float*)d_in[5];
    const float* b_qc     = (const float*)d_in[6];
    const float* W_uqc    = (const float*)d_in[7];
    const float* b_uqc    = (const float*)d_in[8];
    const float* W_att_v  = (const float*)d_in[9];
    const float* W_att_q  = (const float*)d_in[10];
    const float* w_att    = (const float*)d_in[11];
    const float* W_qnet   = (const float*)d_in[12];
    const float* b_qnet   = (const float*)d_in[13];
    const float* W_vnet   = (const float*)d_in[14];
    const float* b_vnet   = (const float*)d_in[15];
    const float* W_ctx    = (const float*)d_in[16];
    const float* b_ctx    = (const float*)d_in[17];
    const float* W_q_na   = (const float*)d_in[18];
    const float* W_k_na   = (const float*)d_in[19];
    const float* W_v_na   = (const float*)d_in[20];
    const float* W_o_na   = (const float*)d_in[21];
    const float* W_cls    = (const float*)d_in[22];
    const float* b_cls    = (const float*)d_in[23];
    const int*   role_ids = (const int*)d_in[24];
    const int*   gt_verb  = (const int*)d_in[25];

    char* ws = (char*)d_ws;
    size_t off = 0;
    auto alloc = [&](size_t bytes) -> void* {
        void* p = ws + off;
        off += (bytes + 255) & ~(size_t)255;
        return p;
    };
    const int BHW = B_ * HW_;   // 6272

    // weights, transposed to [N x K] bf16
    unsigned short* wrefT  = (unsigned short*)alloc((size_t)D_ * C_ * 2);
    unsigned short* wqcT   = (unsigned short*)alloc((size_t)D_ * K_QC * 2);
    unsigned short* wattvT = (unsigned short*)alloc((size_t)D_ * D_ * 2);
    unsigned short* wattqT = (unsigned short*)alloc((size_t)D_ * D_ * 2);
    unsigned short* wqnetT = (unsigned short*)alloc((size_t)D_ * D_ * 2);
    unsigned short* wvnetT = (unsigned short*)alloc((size_t)D_ * D_ * 2);
    unsigned short* wqnaT  = (unsigned short*)alloc((size_t)D_ * D_ * 2);
    unsigned short* wknaT  = (unsigned short*)alloc((size_t)D_ * D_ * 2);
    unsigned short* wvnaT  = (unsigned short*)alloc((size_t)D_ * D_ * 2);
    unsigned short* wonaT  = (unsigned short*)alloc((size_t)D_ * D_ * 2);
    unsigned short* wctxAT = (unsigned short*)alloc((size_t)D_ * D_ * 2);
    unsigned short* wctxBT = (unsigned short*)alloc((size_t)D_ * D_ * 2);
    unsigned short* wuqcT  = (unsigned short*)alloc((size_t)D_ * K_UQC * 2);
    unsigned short* wclsT  = (unsigned short*)alloc((size_t)NCLS_P * D_ * 2);

    // activations
    unsigned short* vT_b    = (unsigned short*)alloc((size_t)BHW * C_ * 2);
    float*          rv_f    = (float*)alloc((size_t)N_ * 2 * E_ * 4);
    unsigned short* rv_b    = (unsigned short*)alloc((size_t)N_ * K_QC * 2);
    float*          ref_f   = (float*)alloc((size_t)BHW * D_ * 4);
    unsigned short* ref_b   = (unsigned short*)alloc((size_t)BHW * D_ * 2);
    float*          qemb_f  = (float*)alloc((size_t)N_ * D_ * 4);
    unsigned short* qemb_b  = (unsigned short*)alloc((size_t)N_ * D_ * 2);
    float*          attv1_f = (float*)alloc((size_t)BHW * D_ * 4);
    float*          attq_f  = (float*)alloc((size_t)N_ * D_ * 4);
    float*          sc1     = (float*)alloc((size_t)N_ * HW_ * 4);
    float*          vemb_f  = (float*)alloc((size_t)N_ * D_ * 4);
    unsigned short* vemb_b  = (unsigned short*)alloc((size_t)N_ * D_ * 2);
    float*          vrepr_f = (float*)alloc((size_t)N_ * D_ * 4);
    float*          qrepr_f = (float*)alloc((size_t)N_ * D_ * 4);
    float*          out0_f  = (float*)alloc((size_t)N_ * D_ * 4);
    unsigned short* cur_b   = (unsigned short*)alloc((size_t)N_ * D_ * 2);
    float*          Qf      = (float*)alloc((size_t)N_ * D_ * 4);
    float*          Kf      = (float*)alloc((size_t)N_ * D_ * 4);
    float*          Vf      = (float*)alloc((size_t)N_ * D_ * 4);
    float*          ctx_f   = (float*)alloc((size_t)N_ * D_ * 4);
    unsigned short* ctx_b   = (unsigned short*)alloc((size_t)N_ * D_ * 2);
    float*          wctxo_f = (float*)alloc((size_t)N_ * D_ * 4);
    unsigned short* wctxo_b = (unsigned short*)alloc((size_t)N_ * D_ * 2);
    float*          gA_f    = (float*)alloc((size_t)N_ * D_ * 4);
    float*          gB_f    = (float*)alloc((size_t)BHW * D_ * 4);
    unsigned short* uimg_b  = (unsigned short*)alloc((size_t)N_ * HW_ * D_ * 2);
    unsigned short* cat_b   = (unsigned short*)alloc((size_t)N_ * K_UQC * 2);
    float*          updq_f  = (float*)alloc((size_t)N_ * D_ * 4);
    unsigned short* updq_b  = (unsigned short*)alloc((size_t)N_ * D_ * 2);
    float*          attv2_f = (float*)alloc((size_t)N_ * HW_ * D_ * 4);
    float*          attq2_f = (float*)alloc((size_t)N_ * D_ * 4);
    float*          sc2     = (float*)alloc((size_t)N_ * HW_ * 4);
    float*          vemb2_f = (float*)alloc((size_t)N_ * D_ * 4);
    unsigned short* vemb2_b = (unsigned short*)alloc((size_t)N_ * D_ * 2);
    float*          vrep2_f = (float*)alloc((size_t)N_ * D_ * 4);
    float*          qrep2_f = (float*)alloc((size_t)N_ * D_ * 4);
    unsigned short* out_b   = (unsigned short*)alloc((size_t)N_ * D_ * 2);
    (void)ws_size; (void)in_sizes; (void)n_in; (void)out_size;

    auto pack = [&](const float* s, unsigned short* d, int sr, int sc, int dr, int dc) {
        int tot = dr * dc;
        k_pack<<<ceilDiv(tot, 256), 256, 0, stream>>>(s, d, sr, sc, dr, dc);
    };
    auto packT = [&](const float* s, unsigned short* d, int sr, int sc, int Kpad, int Ncols) {
        int tot = Ncols * Kpad;
        k_packT<<<ceilDiv(tot, 256), 256, 0, stream>>>(s, d, sr, sc, Kpad, Ncols);
    };
    auto gemm = [&](const unsigned short* A, const unsigned short* BT, const float* bias,
                    float* Cm, int M, int Nn, int K, int ldc, int actN, int epi) {
        dim3 g(Nn / 64, M / 128);
        k_gemm_bf16<<<g, 256, 0, stream>>>(A, BT, bias, Cm, M, Nn, K, ldc, actN, epi);
    };

    // ---- pack weights (transposed, K-major) ----
    packT(W_ref,   wrefT,  C_, D_, C_, D_);
    packT(W_qc,    wqcT,   2 * E_, D_, K_QC, D_);
    packT(W_att_v, wattvT, D_, D_, D_, D_);
    packT(W_att_q, wattqT, D_, D_, D_, D_);
    packT(W_qnet,  wqnetT, D_, D_, D_, D_);
    packT(W_vnet,  wvnetT, D_, D_, D_, D_);
    packT(W_q_na,  wqnaT,  D_, D_, D_, D_);
    packT(W_k_na,  wknaT,  D_, D_, D_, D_);
    packT(W_v_na,  wvnaT,  D_, D_, D_, D_);
    packT(W_o_na,  wonaT,  D_, D_, D_, D_);
    packT(W_ctx,                      wctxAT, D_, D_, D_, D_);
    packT(W_ctx + (size_t)D_ * D_,    wctxBT, D_, D_, D_, D_);
    packT(W_uqc,   wuqcT,  D_ + 2 * E_, D_, K_UQC, D_);
    packT(W_cls,   wclsT,  D_, NLBL, D_, NCLS_P);

    // ---- pack activations ----
    {
        int tot = BHW * C_;
        k_pack_vT<<<ceilDiv(tot, 256), 256, 0, stream>>>(v_org, vT_b, tot);
    }
    {
        int tot = N_ * 2 * E_;
        k_roleverb<<<ceilDiv(tot, 256), 256, 0, stream>>>(verb_emb, role_emb, role_ids, gt_verb, rv_f, tot);
    }
    pack(rv_f, rv_b, N_, 2 * E_, N_, K_QC);

    // ---- img refiner (shared across roles) ----
    gemm(vT_b, wrefT, b_ref, ref_f, BHW, D_, C_, D_, D_, 0);
    pack(ref_f, ref_b, BHW, D_, BHW, D_);

    // ---- q_emb ----
    gemm(rv_b, wqcT, b_qc, qemb_f, N_, D_, K_QC, D_, D_, 0);
    pack(qemb_f, qemb_b, N_, D_, N_, D_);

    // ---- pooled #1 ----
    gemm(ref_b,  wattvT, nullptr, attv1_f, BHW, D_, D_, D_, D_, 0);
    gemm(qemb_b, wattqT, nullptr, attq_f,  N_,  D_, D_, D_, D_, 0);
    k_score<<<N_ * HW_, 256, 0, stream>>>(attv1_f, attq_f, w_att, sc1, 1);
    k_pool<<<N_, 256, 0, stream>>>(sc1, ref_f, (const unsigned short*)nullptr, vemb_f, 1);
    pack(vemb_f, vemb_b, N_, D_, N_, D_);
    gemm(vemb_b, wvnetT, b_vnet, vrepr_f, N_, D_, D_, D_, D_, 1);
    gemm(qemb_b, wqnetT, b_qnet, qrepr_f, N_, D_, D_, D_, D_, 1);
    k_mfb<<<N_, 256, 0, stream>>>(qrepr_f, vrepr_f, out0_f);
    pack(out0_f, cur_b, N_, D_, N_, D_);

    // ---- role self-attention ----
    gemm(cur_b, wqnaT, nullptr, Qf, N_, D_, D_, D_, D_, 0);
    gemm(cur_b, wknaT, nullptr, Kf, N_, D_, D_, D_, D_, 0);
    gemm(cur_b, wvnaT, nullptr, Vf, N_, D_, D_, D_, D_, 0);
    k_roleattn<<<B_ * NH_, 256, 0, stream>>>(Qf, Kf, Vf, ctx_f);
    pack(ctx_f, ctx_b, N_, D_, N_, D_);
    gemm(ctx_b, wonaT, nullptr, wctxo_f, N_, D_, D_, D_, D_, 0);
    pack(wctxo_f, wctxo_b, N_, D_, N_, D_);

    // ---- gating of image regions ----
    gemm(wctxo_b, wctxAT, nullptr, gA_f, N_,  D_, D_, D_, D_, 0);
    gemm(ref_b,   wctxBT, nullptr, gB_f, BHW, D_, D_, D_, D_, 0);
    {
        int tot = N_ * HW_ * D_;
        k_gate<<<ceilDiv(tot, 256), 256, 0, stream>>>(gA_f, gB_f, b_ctx, ref_f, uimg_b, tot);
    }

    // ---- updated query ----
    {
        int tot = N_ * K_UQC;
        k_cat<<<ceilDiv(tot, 256), 256, 0, stream>>>(wctxo_f, rv_f, cat_b, tot);
    }
    gemm(cat_b, wuqcT, b_uqc, updq_f, N_, D_, K_UQC, D_, D_, 0);
    pack(updq_f, updq_b, N_, D_, N_, D_);

    // ---- pooled #2 (per-role image) ----
    gemm(uimg_b, wattvT, nullptr, attv2_f, N_ * HW_, D_, D_, D_, D_, 0);
    gemm(updq_b, wattqT, nullptr, attq2_f, N_, D_, D_, D_, D_, 0);
    k_score<<<N_ * HW_, 256, 0, stream>>>(attv2_f, attq2_f, w_att, sc2, 0);
    k_pool<<<N_, 256, 0, stream>>>(sc2, (const float*)nullptr, uimg_b, vemb2_f, 0);
    pack(vemb2_f, vemb2_b, N_, D_, N_, D_);
    gemm(vemb2_b, wvnetT, b_vnet, vrep2_f, N_, D_, D_, D_, D_, 1);
    gemm(updq_b,  wqnetT, b_qnet, qrep2_f, N_, D_, D_, D_, D_, 1);

    // ---- gated combine + classifier ----
    k_final<<<N_, 256, 0, stream>>>(qrepr_f, qrep2_f, vrep2_f, out0_f, out_b);
    gemm(out_b, wclsT, b_cls, (float*)d_out, N_, NCLS_P, D_, NLBL, NLBL, 0);
}
